// QuantumFeatureMap_70050916598219
// MI455X (gfx1250) — matrix operand order repaired and checked by reading, not compiled, not run on hardware
//
#include <hip/hip_runtime.h>

// ---------------------------------------------------------------------------
// QuantumFeatureMap on gfx1250 (MI455X).
//
// Math: RY-only product state + CNOT-chain permutation folded into parity
// sign masks:
//   out[m][n] = sum_k probs[m][k] * (-1)^popc(k & mask[n])
//   probs[m][k] = prod_q (bit_q(k) ? sin^2 : cos^2)(0.785*x[m][q]), qubit0=MSB
// Implemented as a [16 x 256] x [256 x 48] f16 WMMA GEMM per wave.
// The 256-dim state lives in registers (rank-1 factorized, pk_f16 outer
// product); the constant +/-1 sign matrix is built once per block in LDS.
// ---------------------------------------------------------------------------

typedef __attribute__((ext_vector_type(16))) _Float16     v16h;
typedef __attribute__((ext_vector_type(8)))  float        v8f;
typedef __attribute__((ext_vector_type(2)))  _Float16     h2;
typedef __attribute__((ext_vector_type(4)))  unsigned int u4;

#define NQ   8
#define NOUT 36
#define AH   0.785f /* ALPHA/2 = 1.57/2 */

// Parity masks over k's bits (bit 7-q == qubit q), permutation pre-folded:
// n<8 : qubits 0..n ; n>=8: ZZ(i,j) -> qubits i+1..j ; padded with 12 zeros.
__device__ __constant__ unsigned int kMask[48] = {
  0x80,0xC0,0xE0,0xF0,0xF8,0xFC,0xFE,0xFF,
  0x40,0x60,0x70,0x78,0x7C,0x7E,0x7F,
  0x20,0x30,0x38,0x3C,0x3E,0x3F,
  0x10,0x18,0x1C,0x1E,0x1F,
  0x08,0x0C,0x0E,0x0F,
  0x04,0x06,0x07,
  0x02,0x03,
  0x01,
  0,0,0,0,0,0,0,0,0,0,0,0
};

union Frag {             // one wave-matrix operand, viewed 4 ways
  h2           h[8];     // 8 packed f16 pairs (one per VGPR)
  unsigned int d[8];
  u4           q[2];
  v16h         v;
};

static __device__ __forceinline__ h2 pk2(float a, float b) {
  h2 r; r.x = (_Float16)a; r.y = (_Float16)b; return r;
}

__global__ __launch_bounds__(256)
void qfm_wmma_kernel(const float* __restrict__ x, float* __restrict__ out,
                     int nsamp) {
  // B sign fragments: [frag = nt*8 + chunk][lane][vgpr dword]  (24 KB)
  __shared__ __align__(16) unsigned int Bsh[24][32][8];

  const int lane = threadIdx.x & 31;
  const int wid  = threadIdx.x >> 5;    // 0..7
  const int mrow = lane & 15;           // A row / B,D column lane index
  const int hi   = lane >> 4;           // half-wave (k bit3 = qubit4 for A)

  // ---- cooperative build of the constant sign matrix (3 fragments/wave) ----
  for (int f = wid; f < 24; f += 8) {
    const int nt = f >> 3, c = f & 7;
    const unsigned mask = kMask[nt * 16 + mrow];
    Frag b;
#pragma unroll
    for (int v = 0; v < 8; ++v) {
      // B (32x16) layout: VGPR v, K = j0,j0+1, j0 = 2*(v&3)+16*(v>>2) ; lanes 16-31: +8
      const unsigned k0 = 32u * (unsigned)c + 16u * ((unsigned)v >> 2) + 2u * ((unsigned)v & 3u) + (hi ? 8u : 0u);
      const unsigned p0 = (unsigned)__popc(k0 & mask) & 1u;  // sign of K even
      const unsigned p1 = p0 ^ (mask & 1u);                  // sign of K odd
      b.d[v] = 0x3C003C00u ^ (p0 << 15) ^ (p1 << 31);        // +/-1.0 f16 pair
    }
    *(u4*)&Bsh[f][lane][0] = b.q[0];
    *(u4*)&Bsh[f][lane][4] = b.q[1];
  }
  __syncthreads();

  const int wave = blockIdx.x * 8 + wid;
  const int tile = wave * 16;                       // 16 samples per wave
  if (tile + 16 > nsamp) return;                    // wave-uniform

  // ---- per-sample RY factors (both half-waves load the same row) ----
  const int m = tile + mrow;
  const float4* xr = reinterpret_cast<const float4*>(x + (size_t)m * NQ);
  const float4 xa = xr[0], xb = xr[1];
  const float xf[NQ] = {xa.x, xa.y, xa.z, xa.w, xb.x, xb.y, xb.z, xb.w};

  float c2[NQ], s2[NQ];
#pragma unroll
  for (int q = 0; q < NQ; ++q) {
    const float c = __cosf(AH * xf[q]);
    c2[q] = c * c;
    s2[q] = 1.0f - c2[q];
  }
  auto F = [&](int q, int b) { return b ? s2[q] : c2[q]; };

  // c3[c]: product over qubits 0..2 (c == k>>5 == K-chunk index)
  float c01[4], c3[8];
#pragma unroll
  for (int a = 0; a < 4; ++a) c01[a] = F(0, (a >> 1) & 1) * F(1, a & 1);
#pragma unroll
  for (int c = 0; c < 8; ++c) c3[c] = c01[c >> 1] * F(2, c & 1);

  // r16[j]: product over qubits 3,5,6,7 with qubit4 fixed to `hi`
  // j: bit3=q3 (k bit4), bit2=q5 (k bit2), bit1=q6 (k bit1), bit0=q7 (k bit0)
  float f34[2], f345[4], u67[4], r16[16];
#pragma unroll
  for (int a = 0; a < 2; ++a) f34[a] = F(3, a) * F(4, hi);
#pragma unroll
  for (int a = 0; a < 4; ++a) f345[a] = f34[(a >> 1) & 1] * F(5, a & 1);
#pragma unroll
  for (int a = 0; a < 4; ++a) u67[a] = F(6, (a >> 1) & 1) * F(7, a & 1);
#pragma unroll
  for (int j = 0; j < 16; ++j) r16[j] = f345[j >> 2] * u67[j & 3];

  // Packed f16 factor tables: A fragment dword (c,v) = ch[c] * rh[v]
  // A (16x32) layout per lane: VGPR v pair K = {j0, j0+1}, j0 = 2*(v&3)+16*(v>>2) ; lanes 16-31: +8
  // -> rh[v] = (r16[2v], r16[2v+1]) after the j0 remap below.
  h2 ch[8], rh[8];
#pragma unroll
  for (int c = 0; c < 8; ++c) ch[c] = pk2(c3[c], c3[c]);
#pragma unroll
  for (int v = 0; v < 8; ++v) {
    const int j0 = ((v >> 2) << 3) | ((v & 3) << 1);
    rh[v] = pk2(r16[j0], r16[j0 | 1]);
  }

  // ---- all 8 A fragments via v_pk_mul_f16 outer product ----
  Frag af[8];
#pragma unroll
  for (int c = 0; c < 8; ++c)
#pragma unroll
    for (int v = 0; v < 8; ++v)
      af[c].h[v] = ch[c] * rh[v];

  // ---- GEMM: 3 N-tiles x 8 K-chunks of WMMA, B streamed from LDS ----
#pragma unroll
  for (int nt = 0; nt < 3; ++nt) {
    v8f acc = {0.f, 0.f, 0.f, 0.f, 0.f, 0.f, 0.f, 0.f};
#pragma unroll
    for (int c = 0; c < 8; ++c) {
      Frag bf;
      bf.q[0] = *(const u4*)&Bsh[nt * 8 + c][lane][0];
      bf.q[1] = *(const u4*)&Bsh[nt * 8 + c][lane][4];
      acc = __builtin_amdgcn_wmma_f32_16x16x32_f16(
          /*neg_a=*/false, af[c].v, /*neg_b=*/false, bf.v,
          /*c_mod=*/(short)0, acc, /*reuse_a=*/false, /*reuse_b=*/false);
    }
    // D layout: VGPR r -> row (r + 8*hi), col = lane&15
    const int n = nt * 16 + mrow;
#pragma unroll
    for (int r = 0; r < 8; ++r) {
      if (nt < 2 || mrow < NOUT - 32)   // n < 36; uniform for nt=0,1
        out[(size_t)(tile + r + (hi ? 8 : 0)) * NOUT + n] = acc[r];
    }
  }
}

extern "C" void kernel_launch(void* const* d_in, const int* in_sizes, int n_in,
                              void* d_out, int out_size, void* d_ws, size_t ws_size,
                              hipStream_t stream) {
  const float* x = (const float*)d_in[0];
  float* out = (float*)d_out;
  const int nsamp = in_sizes[0] / NQ;             // 131072
  const int samples_per_block = 8 * 16;           // 8 waves * 16 rows
  const int blocks = (nsamp + samples_per_block - 1) / samples_per_block;
  qfm_wmma_kernel<<<blocks, 256, 0, stream>>>(x, out, nsamp);
}
